// UltraMinimalNeuralODE_43765716746783
// MI455X (gfx1250) — matrix operand order repaired and checked by reading, not compiled, not run on hardware
//
#include <hip/hip_runtime.h>
#include <hip/hip_bf16.h>
#include <math.h>

// ---------------- problem constants (from reference) ----------------
#define B_    4096
#define S_    256
#define D_    18      // STATE_DIM
#define H_    16      // HIDDEN_DIM
#define L_    3       // LATENT_DIM
#define T_    64
#define EPS_  1e-5f

#define NROWS   ((long long)B_ * S_)          // 1,048,576 rows through encode/decode
#define WAVES_PER_BLOCK 8
#define ROWS_PER_WAVE   32                    // two 16-row WMMA tiles per wave
#define GRID_A  (int)(NROWS / (WAVES_PER_BLOCK * ROWS_PER_WAVE))   // 4096

typedef __attribute__((ext_vector_type(16))) _Float16 v16h;
typedef __attribute__((ext_vector_type(8)))  float    v8f;

// =====================================================================
// Kernel A: fused encode (WMMA 16x16x32 f16) + BN/ReLU + z projection +
//           decode + writes of x_recon and z_seq.  One pass over x.
// =====================================================================
__global__ __launch_bounds__(256) void fused_encode_decode(
    const float* __restrict__ x,                                  // [B*S, 18]
    const float* __restrict__ ew1, const float* __restrict__ eb1, // enc 18->16
    const float* __restrict__ ebg, const float* __restrict__ ebb,
    const float* __restrict__ ebm, const float* __restrict__ ebv,
    const float* __restrict__ ew2, const float* __restrict__ eb2, // enc 16->3
    const float* __restrict__ dw1, const float* __restrict__ db1, // dec 3->16
    const float* __restrict__ dbg, const float* __restrict__ dbb,
    const float* __restrict__ dbm, const float* __restrict__ dbv,
    const float* __restrict__ dw2, const float* __restrict__ db2, // dec 16->18
    float* __restrict__ x_recon,                                  // [B*S, 18]
    float* __restrict__ z_seq)                                    // [B*S, 3]
{
    // per-wave h tiles: 32 rows x 16 cols, stride 17 -> conflict-free
    __shared__ float s_htile[WAVES_PER_BLOCK][ROWS_PER_WAVE][17];
    __shared__ float s_ew2[L_ * H_];   // 48
    __shared__ float s_eb2[L_];
    __shared__ float s_dw1[H_ * L_];   // 48
    __shared__ float s_db1[H_];
    __shared__ float s_dw2s[D_ * H_];  // dec_w2 with BN scale folded in (288)
    __shared__ float s_db2f[D_];       // dec_b2 with BN bias folded in

    const int tid = threadIdx.x;

    // ---- one-time per-block constant preload + BN folding ----
    if (tid < L_ * H_) { s_ew2[tid] = ew2[tid]; s_dw1[tid] = dw1[tid]; }
    if (tid < L_)       s_eb2[tid] = eb2[tid];
    if (tid < H_)       s_db1[tid] = db1[tid];
    if (tid < D_ * H_) {                       // fold BN scale into dec_w2
        int n = tid % H_;
        float sc = dbg[n] * rsqrtf(dbv[n] + EPS_);
        s_dw2s[tid] = dw2[tid] * sc;
    }
    if (tid < D_) {                            // fold BN bias through dec_w2
        float acc = db2[tid];
        for (int n = 0; n < H_; ++n) {
            float sc = dbg[n] * rsqrtf(dbv[n] + EPS_);
            float bi = dbb[n] - dbm[n] * sc;
            acc += bi * dw2[tid * H_ + n];
        }
        s_db2f[tid] = acc;
    }
    __syncthreads();

    const int wave = tid >> 5;
    const int lane = tid & 31;
    const long long row0 =
        ((long long)blockIdx.x * WAVES_PER_BLOCK + wave) * ROWS_PER_WAVE;

    // ---- B matrix: enc_w1^T (K=18 padded to 32) in the 16-bit B layout ----
    // lane n%16 owns output column n; lanes 0-15 K=0..7,16..23, lanes 16-31 K=8..15,24..31
    const int  ncol = lane & 15;
    const int  kb_b = (lane < 16) ? 0 : 8;
    const bool lo   = (lane < 16);
    v16h Bm;
#pragma unroll
    for (int i = 0; i < 16; ++i) {
        int k = (i < 8) ? (kb_b + i) : (16 + kb_b + (i - 8));
        Bm[i] = (_Float16)((k < D_) ? ew1[ncol * D_ + k] : 0.f);
    }
    // encoder BN (column = ncol is a lane constant)
    const float esc  = ebg[ncol] * rsqrtf(ebv[ncol] + EPS_);
    const float ebi  = ebb[ncol] - ebm[ncol] * esc;
    const float eb1c = eb1[ncol];

    // ---- two 16x16x32 WMMA tiles (rows row0..row0+31) ----
    const int mrow = lane & 15;
    const int kb_a = lo ? 0 : 8;               // 16-bit A layout K split
#pragma unroll
    for (int t = 0; t < 2; ++t) {
        const float* xr = x + (row0 + t * 16 + mrow) * D_;
        // prefetch the next wave-chunk's row (speculative, gfx1250 path)
        __builtin_prefetch(xr + (long long)ROWS_PER_WAVE * WAVES_PER_BLOCK * D_, 0, 0);

        // Branch-free A-tile build: every lane loads its contiguous 8 floats
        // plus x[16],x[17] (same cacheline); lanes>=16 select 0 via cndmask
        // instead of taking a divergent exec-mask branch.
        float t16 = xr[16];
        float t17 = xr[17];
        v16h Am;
#pragma unroll
        for (int i = 0; i < 8; ++i) {          // K = kb_a .. kb_a+7 (always < 16)
            Am[i] = (_Float16)xr[kb_a + i];
        }
        Am[8] = lo ? (_Float16)t16 : (_Float16)0.f;   // K=16 (lo) / K=24 pad (hi)
        Am[9] = lo ? (_Float16)t17 : (_Float16)0.f;   // K=17 (lo) / K=25 pad (hi)
#pragma unroll
        for (int i = 10; i < 16; ++i) Am[i] = (_Float16)0.f;  // K pad >= 18

        v8f Cm = {};
        Cm = __builtin_amdgcn_wmma_f32_16x16x32_f16(
                 /*neg_a=*/false, Am, /*neg_b=*/false, Bm,
                 /*c_mod=*/(short)0, Cm, /*reuse_a=*/false, /*reuse_b=*/false);

        // bias + ReLU + BN, scatter C layout -> row-major LDS tile
#pragma unroll
        for (int r = 0; r < 8; ++r) {
            int M = r + (lo ? 0 : 8);
            float h = Cm[r] + eb1c;
            h = fmaxf(h, 0.f);
            h = h * esc + ebi;
            s_htile[wave][t * 16 + M][ncol] = h;
        }
    }
    __syncthreads();

    // ---- per-lane row tail: z = h@ew2^T, decode, write outputs ----
    const float* hrow = &s_htile[wave][lane][0];
    float z[L_];
#pragma unroll
    for (int j = 0; j < L_; ++j) {
        float acc = s_eb2[j];
#pragma unroll
        for (int n = 0; n < H_; ++n) acc += hrow[n] * s_ew2[j * H_ + n];
        z[j] = acc;
    }
    const long long grow = row0 + lane;
    float* zp = z_seq + grow * L_;
    zp[0] = z[0]; zp[1] = z[1]; zp[2] = z[2];

    float h2[H_];
#pragma unroll
    for (int n = 0; n < H_; ++n) {
        float a = s_db1[n] + z[0] * s_dw1[n * 3] + z[1] * s_dw1[n * 3 + 1]
                           + z[2] * s_dw1[n * 3 + 2];
        h2[n] = fmaxf(a, 0.f);                 // BN folded into s_dw2s / s_db2f
    }
    // row stride is 72B -> 8B aligned; emit as 9 float2 (global_store_b64)
    float2* xo2 = (float2*)(x_recon + grow * D_);
#pragma unroll
    for (int p = 0; p < D_ / 2; ++p) {
        float acc0 = s_db2f[2 * p];
        float acc1 = s_db2f[2 * p + 1];
#pragma unroll
        for (int n = 0; n < H_; ++n) {
            acc0 += h2[n] * s_dw2s[(2 * p) * H_ + n];
            acc1 += h2[n] * s_dw2s[(2 * p + 1) * H_ + n];
        }
        xo2[p] = make_float2(acc0, acc1);
    }
}

// =====================================================================
// Kernel B: per-batch z_mean + classifier + sequential Euler ODE,
//           then parallel decode of the 64 trajectory points -> x_pred.
// =====================================================================
__global__ __launch_bounds__(64) void ode_cls_pred(
    const float* __restrict__ z_seq,                              // [B,S,3]
    const float* __restrict__ dw1, const float* __restrict__ db1,
    const float* __restrict__ dbg, const float* __restrict__ dbb,
    const float* __restrict__ dbm, const float* __restrict__ dbv,
    const float* __restrict__ dw2, const float* __restrict__ db2,
    const float* __restrict__ ow1, const float* __restrict__ ob1, // ode 3->8->3
    const float* __restrict__ ow2, const float* __restrict__ ob2,
    const float* __restrict__ cw1, const float* __restrict__ cb1, // cls 3->4->1
    const float* __restrict__ cw2, const float* __restrict__ cb2,
    float* __restrict__ x_pred,                                   // [B,T,18]
    float* __restrict__ prob)                                     // [B]
{
    __shared__ float s_part[64][3];
    __shared__ float s_traj[T_][3];

    const int b = blockIdx.x;
    const int t = threadIdx.x;
    const float* zb = z_seq + (long long)b * S_ * L_;

    // z_mean reduction over S=256
    float a0 = 0.f, a1 = 0.f, a2 = 0.f;
    for (int s = t; s < S_; s += 64) {
        a0 += zb[s * 3]; a1 += zb[s * 3 + 1]; a2 += zb[s * 3 + 2];
    }
    s_part[t][0] = a0; s_part[t][1] = a1; s_part[t][2] = a2;
    __syncthreads();

    if (t == 0) {
        float m0 = 0.f, m1 = 0.f, m2 = 0.f;
        for (int i = 0; i < 64; ++i) {
            m0 += s_part[i][0]; m1 += s_part[i][1]; m2 += s_part[i][2];
        }
        m0 /= (float)S_; m1 /= (float)S_; m2 /= (float)S_;

        // classifier
        float hc[4];
        for (int j = 0; j < 4; ++j)
            hc[j] = fmaxf(cb1[j] + m0 * cw1[j * 3] + m1 * cw1[j * 3 + 1]
                                 + m2 * cw1[j * 3 + 2], 0.f);
        float lg = cb2[0];
        for (int j = 0; j < 4; ++j) lg += hc[j] * cw2[j];
        prob[b] = 1.f / (1.f + expf(-lg));

        // sequential Euler integration from z0 = z_seq[b, S-1]
        float z0v = zb[(S_ - 1) * 3];
        float z1v = zb[(S_ - 1) * 3 + 1];
        float z2v = zb[(S_ - 1) * 3 + 2];
        for (int step = 0; step < T_; ++step) {
            float hh[8];
            for (int j = 0; j < 8; ++j)
                hh[j] = tanhf(ob1[j] + z0v * ow1[j * 3] + z1v * ow1[j * 3 + 1]
                                     + z2v * ow1[j * 3 + 2]);
            float f0 = ob2[0], f1 = ob2[1], f2 = ob2[2];
            for (int j = 0; j < 8; ++j) {
                f0 += hh[j] * ow2[j];
                f1 += hh[j] * ow2[8 + j];
                f2 += hh[j] * ow2[16 + j];
            }
            z0v += 0.1f * f0; z1v += 0.1f * f1; z2v += 0.1f * f2;
            s_traj[step][0] = z0v; s_traj[step][1] = z1v; s_traj[step][2] = z2v;
        }
    }
    __syncthreads();

    // decode trajectory point t -> x_pred[b, t, :]
    const float zt0 = s_traj[t][0], zt1 = s_traj[t][1], zt2 = s_traj[t][2];
    float h2[H_];
#pragma unroll
    for (int n = 0; n < H_; ++n) {
        float sc = dbg[n] * rsqrtf(dbv[n] + EPS_);
        float bi = dbb[n] - dbm[n] * sc;
        float a  = fmaxf(db1[n] + zt0 * dw1[n * 3] + zt1 * dw1[n * 3 + 1]
                                 + zt2 * dw1[n * 3 + 2], 0.f);
        h2[n] = a * sc + bi;
    }
    // 72B row stride -> 8B aligned; 9 float2 stores
    float2* xo2 = (float2*)(x_pred + ((long long)b * T_ + t) * D_);
#pragma unroll
    for (int p = 0; p < D_ / 2; ++p) {
        float acc0 = db2[2 * p];
        float acc1 = db2[2 * p + 1];
#pragma unroll
        for (int n = 0; n < H_; ++n) {
            acc0 += h2[n] * dw2[(2 * p) * H_ + n];
            acc1 += h2[n] * dw2[(2 * p + 1) * H_ + n];
        }
        xo2[p] = make_float2(acc0, acc1);
    }
}

// =====================================================================
extern "C" void kernel_launch(void* const* d_in, const int* in_sizes, int n_in,
                              void* d_out, int out_size, void* d_ws, size_t ws_size,
                              hipStream_t stream) {
    const float* x   = (const float*)d_in[0];
    // d_in[1] = t_span (unused: dt is fixed 0.1 and T=64 in the reference)
    const float* ew1 = (const float*)d_in[2];
    const float* eb1 = (const float*)d_in[3];
    const float* ebg = (const float*)d_in[4];
    const float* ebb = (const float*)d_in[5];
    const float* ebm = (const float*)d_in[6];
    const float* ebv = (const float*)d_in[7];
    const float* ew2 = (const float*)d_in[8];
    const float* eb2 = (const float*)d_in[9];
    const float* dw1 = (const float*)d_in[10];
    const float* db1 = (const float*)d_in[11];
    const float* dbg = (const float*)d_in[12];
    const float* dbb = (const float*)d_in[13];
    const float* dbm = (const float*)d_in[14];
    const float* dbv = (const float*)d_in[15];
    const float* dw2 = (const float*)d_in[16];
    const float* db2 = (const float*)d_in[17];
    const float* ow1 = (const float*)d_in[18];
    const float* ob1 = (const float*)d_in[19];
    const float* ow2 = (const float*)d_in[20];
    const float* ob2 = (const float*)d_in[21];
    const float* cw1 = (const float*)d_in[22];
    const float* cb1 = (const float*)d_in[23];
    const float* cw2 = (const float*)d_in[24];
    const float* cb2 = (const float*)d_in[25];

    float* out      = (float*)d_out;
    float* x_recon  = out;                                        // B*S*18
    float* x_pred   = x_recon + (size_t)B_ * S_ * D_;             // B*T*18
    float* prob     = x_pred  + (size_t)B_ * T_ * D_;             // B
    float* z_seq    = prob    + (size_t)B_;                       // B*S*3

    fused_encode_decode<<<GRID_A, 256, 0, stream>>>(
        x, ew1, eb1, ebg, ebb, ebm, ebv, ew2, eb2,
        dw1, db1, dbg, dbb, dbm, dbv, dw2, db2,
        x_recon, z_seq);

    ode_cls_pred<<<B_, 64, 0, stream>>>(
        z_seq, dw1, db1, dbg, dbb, dbm, dbv, dw2, db2,
        ow1, ob1, ow2, ob2, cw1, cb1, cw2, cb2,
        x_pred, prob);
}
